// LASCC_7370163880170
// MI455X (gfx1250) — matrix operand order repaired and checked by reading, not compiled, not run on hardware
//
#include <hip/hip_runtime.h>
#include <hip/hip_bf16.h>

typedef __attribute__((ext_vector_type(16))) _Float16 v16h;
typedef __attribute__((ext_vector_type(8)))  float    v8f;

#define BATCH   8
#define CHN     128
#define NPATCH  1024
#define HW      64

union FragH { int4 q[2]; v16h h; };

// -------------------------------------------------------------------------
// Kernel 1: per-pixel L2 normalize over channels, pack diagonal-pixel rows
// xd[b][p][n][c] as f16.  One thread per (b,p,n).
// -------------------------------------------------------------------------
__global__ void __launch_bounds__(256)
norm_pack_kernel(const float* __restrict__ x, _Float16* __restrict__ xd)
{
    int tid = blockIdx.x * 256 + threadIdx.x;      // 8*2*1024 = 16384 threads
    int b = tid >> 11;
    int p = (tid >> 10) & 1;
    int n = tid & 1023;
    int i = n >> 5, j = n & 31;
    int r = 2 * i + p, c = 2 * j + p;              // diagonal pixel of patch n
    const float* xb = x + (size_t)b * CHN * 4096 + (r * HW + c);
    float ss = 0.f;
    #pragma unroll 8
    for (int ch = 0; ch < CHN; ++ch) { float v = xb[(size_t)ch * 4096]; ss += v * v; }
    float inv = 1.0f / fmaxf(sqrtf(ss), 1e-12f);
    _Float16* dst = xd + ((size_t)(b * 2 + p) * NPATCH + n) * CHN;
    #pragma unroll 8
    for (int ch = 0; ch < CHN; ++ch) dst[ch] = (_Float16)(xb[(size_t)ch * 4096] * inv);
}

// -------------------------------------------------------------------------
// Kernel 2: corr = Xd * Xd^T per (b,p) via v_wmma_f32_16x16x32_f16, fused
// with mask * alpha epilogue; writes a[b][3][1024][1024] fp32.
// Block = 256 threads (8 waves), block tile 128x64, wave tile 32x32 per p.
// -------------------------------------------------------------------------
__global__ void __launch_bounds__(256)
corr_gemm_kernel(const _Float16* __restrict__ xd, const float* __restrict__ alpha_p,
                 float* __restrict__ aBuf)
{
    __shared__ __align__(16) _Float16 As[2][128 * 32];   // row-block rows, 32 k
    __shared__ __align__(16) _Float16 Bs[2][64 * 32];    // col-block rows, 32 k

    const int b       = blockIdx.z;
    const int rowBase = blockIdx.y * 128;
    const int colBase = blockIdx.x * 64;
    const int tid  = threadIdx.x;
    const int lane = tid & 31;
    const int warp = tid >> 5;
    const int wr = warp >> 1, wc = warp & 1;      // 4x2 wave grid

    const _Float16* xdb = xd + (size_t)b * 2 * NPATCH * CHN;

    v8f zero = {};
    v8f acc[2][2][2];
    #pragma unroll
    for (int p = 0; p < 2; ++p)
        #pragma unroll
        for (int mi = 0; mi < 2; ++mi)
            #pragma unroll
            for (int ni = 0; ni < 2; ++ni) acc[p][mi][ni] = zero;

    for (int kk = 0; kk < CHN; kk += 32) {
        // ---- stage tiles into LDS (16B chunks, fully coalesced) ----
        for (int t = tid; t < 1536; t += 256) {
            if (t < 1024) {                               // As: 2p * 128 rows * 4 chunks
                int p = t >> 9, cc = t & 511, row = cc >> 2, q = cc & 3;
                const int4* src = (const int4*)(xdb + (size_t)p * NPATCH * CHN
                                  + (size_t)(rowBase + row) * CHN + kk + q * 8);
                ((int4*)&As[p][row * 32])[q] = *src;
            } else {                                      // Bs: 2p * 64 rows * 4 chunks
                int t2 = t - 1024;
                int p = t2 >> 8, cc = t2 & 255, row = cc >> 2, q = cc & 3;
                const int4* src = (const int4*)(xdb + (size_t)p * NPATCH * CHN
                                  + (size_t)(colBase + row) * CHN + kk + q * 8);
                ((int4*)&Bs[p][row * 32])[q] = *src;
            }
        }
        __syncthreads();

        const int lm = lane & 15;
        const int ka = (lane < 16) ? 0 : 8;    // A f16 layout: lanes0-15 K0-7/16-23
        const int kb = (lane < 16) ? 0 : 8;    // B f16 layout: lanes0-15 K0-7/16-23

        #pragma unroll
        for (int p = 0; p < 2; ++p) {
            FragH fa[2], fb[2];
            #pragma unroll
            for (int mi = 0; mi < 2; ++mi) {
                const _Float16* ab = &As[p][(wr * 32 + mi * 16 + lm) * 32];
                fa[mi].q[0] = *(const int4*)(ab + ka);
                fa[mi].q[1] = *(const int4*)(ab + ka + 16);
            }
            #pragma unroll
            for (int ni = 0; ni < 2; ++ni) {
                const _Float16* bb = &Bs[p][(wc * 32 + ni * 16 + lm) * 32];
                fb[ni].q[0] = *(const int4*)(bb + kb);
                fb[ni].q[1] = *(const int4*)(bb + kb + 16);
            }
            #pragma unroll
            for (int mi = 0; mi < 2; ++mi)
                #pragma unroll
                for (int ni = 0; ni < 2; ++ni)
                    acc[p][mi][ni] = __builtin_amdgcn_wmma_f32_16x16x32_f16(
                        false, fa[mi].h, false, fb[ni].h,
                        (short)0, acc[p][mi][ni], false, false);
        }
        __syncthreads();
    }

    // ---- epilogue: a = alpha * (1 - gauss) * corr, 3 channels ----
    const float alphav = alpha_p[0];
    const size_t chStride = (size_t)NPATCH * NPATCH;
    float* ab0 = aBuf + (size_t)b * 3 * chStride;
    #pragma unroll
    for (int mi = 0; mi < 2; ++mi)
        #pragma unroll
        for (int ni = 0; ni < 2; ++ni)
            #pragma unroll
            for (int j = 0; j < 8; ++j) {
                int gm = rowBase + wr * 32 + mi * 16 + j + ((lane >> 4) << 3);
                int gn = colBase + wc * 32 + ni * 16 + (lane & 15);
                float dr = (float)((gm >> 5) - (gn >> 5));
                float dc = (float)((gm & 31) - (gn & 31));
                // sr=sc=0.05*32=1.6 -> 2*sr^2 = 5.12
                float w = alphav * (1.0f - __expf(-(dr * dr + dc * dc) * (1.0f / 5.12f)));
                float c0 = acc[0][mi][ni][j];
                float c1 = acc[1][mi][ni][j];
                size_t o = (size_t)gm * NPATCH + gn;
                ab0[o]                = w * c0;
                ab0[o + chStride]     = w * c1;
                ab0[o + 2 * chStride] = w * 0.5f * (c0 + c1);
            }
}

// -------------------------------------------------------------------------
// Kernel 3: per-row max & sum(exp) (a is symmetric -> col stats == row stats)
// One wave per row; row cached in 32 VGPRs.
// -------------------------------------------------------------------------
__global__ void __launch_bounds__(256)
row_stats_kernel(const float* __restrict__ aBuf,
                 float* __restrict__ rmax, float* __restrict__ rsum)
{
    int row  = blockIdx.x * 8 + (threadIdx.x >> 5);   // 24576 rows
    int lane = threadIdx.x & 31;
    const float* ar = aBuf + (size_t)row * NPATCH;
    float v[32];
    float m = -3.402823466e38f;
    #pragma unroll
    for (int i = 0; i < 32; ++i) { v[i] = ar[lane + i * 32]; m = fmaxf(m, v[i]); }
    #pragma unroll
    for (int off = 16; off > 0; off >>= 1) m = fmaxf(m, __shfl_xor(m, off, 32));
    float s = 0.f;
    #pragma unroll
    for (int i = 0; i < 32; ++i) s += __expf(v[i] - m);
    #pragma unroll
    for (int off = 16; off > 0; off >>= 1) s += __shfl_xor(s, off, 32);
    if (lane == 0) { rmax[row] = m; rsum[row] = s; }
}

// -------------------------------------------------------------------------
// Kernel 4: s[n,m] = exp(2a - rm[n] - rm[m]) / (rs[n]*rs[m]); top-3 over m
// per (b,ch,n); scatter to pixels of patch n by channel.
// -------------------------------------------------------------------------
__device__ __forceinline__ void ins3(float x, float& t0, float& t1, float& t2)
{
    if (x > t2) {
        if (x > t1) { t2 = t1; if (x > t0) { t1 = t0; t0 = x; } else t1 = x; }
        else t2 = x;
    }
}

__global__ void __launch_bounds__(256)
topk_kernel(const float* __restrict__ aBuf, const float* __restrict__ rmax,
            const float* __restrict__ rsum, float* __restrict__ out)
{
    int row  = blockIdx.x * 8 + (threadIdx.x >> 5);   // row = (b*3+ch)*1024 + n
    int lane = threadIdx.x & 31;
    int n     = row & 1023;
    int bch   = row >> 10;                            // b*3+ch
    int b     = bch / 3;
    int ch    = bch - b * 3;
    int sbase = row - n;                              // stats base for this (b,ch)

    const float* ar = aBuf + (size_t)row * NPATCH;
    float rm_n = rmax[row];
    float inv_rs_n = 1.0f / rsum[row];

    float t0 = -1e30f, t1 = -1e30f, t2 = -1e30f;
    for (int i = 0; i < 32; ++i) {
        int m = lane + i * 32;
        float val = __expf(2.0f * ar[m] - rm_n - rmax[sbase + m])
                    * inv_rs_n / rsum[sbase + m];
        ins3(val, t0, t1, t2);
    }
    #pragma unroll
    for (int off = 16; off > 0; off >>= 1) {
        float a0 = __shfl_xor(t0, off, 32);
        float a1 = __shfl_xor(t1, off, 32);
        float a2 = __shfl_xor(t2, off, 32);
        ins3(a0, t0, t1, t2); ins3(a1, t0, t1, t2); ins3(a2, t0, t1, t2);
    }
    if (lane == 0) {
        int i = n >> 5, j = n & 31;
        int r0 = 2 * i, c0 = 2 * j;
        float tv[3] = { t0, t1, t2 };
        #pragma unroll
        for (int k = 0; k < 3; ++k) {
            float* ob = out + (size_t)(b * 3 + k) * HW * HW;
            if (ch == 0)      ob[r0 * HW + c0] = tv[k];              // pixel (even,even)
            else if (ch == 1) ob[(r0 + 1) * HW + c0 + 1] = tv[k];    // pixel (odd,odd)
            else { ob[r0 * HW + c0 + 1] = tv[k];                     // avg -> off-diag
                   ob[(r0 + 1) * HW + c0] = tv[k]; }
        }
    }
}

// -------------------------------------------------------------------------
extern "C" void kernel_launch(void* const* d_in, const int* in_sizes, int n_in,
                              void* d_out, int out_size, void* d_ws, size_t ws_size,
                              hipStream_t stream)
{
    const float* x     = (const float*)d_in[0];
    const float* alpha = (const float*)d_in[1];
    float* out = (float*)d_out;

    char* ws = (char*)d_ws;
    const size_t A_BYTES  = (size_t)BATCH * 3 * NPATCH * NPATCH * 4;   // 96 MB
    const size_t XD_BYTES = (size_t)BATCH * 2 * NPATCH * CHN * 2;      // 4 MB
    float*    aBuf = (float*)ws;
    _Float16* xd   = (_Float16*)(ws + A_BYTES);
    float*    rmax = (float*)(ws + A_BYTES + XD_BYTES);
    float*    rsum = rmax + BATCH * 3 * NPATCH;

    norm_pack_kernel<<<64, 256, 0, stream>>>(x, xd);

    dim3 g(NPATCH / 64, NPATCH / 128, BATCH);   // (16, 8, 8)
    corr_gemm_kernel<<<g, 256, 0, stream>>>(xd, alpha, aBuf);

    row_stats_kernel<<<BATCH * 3 * NPATCH / 8, 256, 0, stream>>>(aBuf, rmax, rsum);
    topk_kernel<<<BATCH * 3 * NPATCH / 8, 256, 0, stream>>>(aBuf, rmax, rsum, out);
}
